// HybridMambaBlock_70703751627062
// MI455X (gfx1250) — matrix operand order repaired and checked by reading, not compiled, not run on hardware
//
#include <hip/hip_runtime.h>
#include <math.h>

// ---------------- types ----------------
typedef __bf16 bf16;
typedef __attribute__((ext_vector_type(16))) __bf16 v16bf;
typedef __attribute__((ext_vector_type(8)))  __bf16 v8bf;
typedef __attribute__((ext_vector_type(8)))  float  v8f;

union V16U { v16bf v; v8bf h[2]; };

#define B_DIM   2
#define L_DIM   2048
#define DMODEL  1024
#define DINNER  2048
#define DSTATE  16
#define DTRANK  64
#define TOKENS  (B_DIM * L_DIM)   // 4096

static __device__ __forceinline__ v8f wmma_bf16(v16bf a, v16bf b, v8f c) {
    // D = A(16x32) * B(32x16) + C, fp32 accumulate
    return __builtin_amdgcn_wmma_f32_16x16x32_bf16(false, a, false, b, (short)0, c, false, false);
}

// ---------------- fp32 -> bf16 convert ----------------
__global__ __launch_bounds__(256) void k_cvt_bf16(const float* __restrict__ s,
                                                  bf16* __restrict__ d, int n) {
    int i = blockIdx.x * 256 + threadIdx.x;
    if (i < n) d[i] = (bf16)s[i];
}

// ---------------- RMSNorm -> bf16 ----------------
__global__ __launch_bounds__(256) void k_rmsnorm(const float* __restrict__ hs,
                                                 const float* __restrict__ w,
                                                 bf16* __restrict__ out) {
    __shared__ float sm[256];
    const int t = blockIdx.x;                       // token
    const float* row = hs + (size_t)t * DMODEL;
    const int i = threadIdx.x;
    float x0 = row[i], x1 = row[i + 256], x2 = row[i + 512], x3 = row[i + 768];
    sm[i] = x0 * x0 + x1 * x1 + x2 * x2 + x3 * x3;
    __syncthreads();
    for (int off = 128; off > 0; off >>= 1) {
        if (i < off) sm[i] += sm[i + off];
        __syncthreads();
    }
    const float r = rsqrtf(sm[0] * (1.0f / DMODEL) + 1e-5f);
    bf16* orow = out + (size_t)t * DMODEL;
    orow[i]       = (bf16)(x0 * r * w[i]);
    orow[i + 256] = (bf16)(x1 * r * w[i + 256]);
    orow[i + 512] = (bf16)(x2 * r * w[i + 512]);
    orow[i + 768] = (bf16)(x3 * r * w[i + 768]);
}

// ---------------- WMMA GEMM: out[M,N] = A[M,K] * W[N,K]^T ----------------
// Each wave computes a (16*MT)x(16*NT) tile with MT*NT v_wmma per 32-wide K-step.
// EPI 0: plain fp32 store           (outF = dst)
// EPI 1: x_proj split: col<64 -> bf16 dt_raw (outBF), col>=64 -> fp32 B/C (outF)
// EPI 2: dt: softplus(acc + bias)   (auxF = bias, outF = dt)
// EPI 3: out_proj: residual + acc   (auxF = residual, outF = d_out)
template <int EPI, int MT, int NT>
__global__ __launch_bounds__(256)
void k_gemm(const bf16* __restrict__ A, const bf16* __restrict__ W,
            int M, int N, int K,
            float* __restrict__ outF, bf16* __restrict__ outBF,
            const float* __restrict__ auxF) {
    const int lane = threadIdx.x & 31;
    const int wid  = (blockIdx.x * 256 + threadIdx.x) >> 5;
    const int nStrips = N / (16 * NT);
    const int mTile  = wid / nStrips;
    const int nStrip = wid % nStrips;
    if (mTile * (16 * MT) >= M) return;

    const int q = lane & 15;                       // row (A) / W row within tile
    const int h = lane >> 4;                       // half selector

    const bf16* aP[MT];
    const bf16* wP[NT];
#pragma unroll
    for (int mt = 0; mt < MT; ++mt)
        aP[mt] = A + (size_t)(mTile * (16 * MT) + mt * 16 + q) * K;
#pragma unroll
    for (int nt = 0; nt < NT; ++nt)
        wP[nt] = W + (size_t)(nStrip * (16 * NT) + nt * 16 + q) * K + 16 * h;

    v8f acc[MT][NT] = {};

    for (int k0 = 0; k0 < K; k0 += 32) {
        V16U  fa[MT];
        v16bf fb[NT];
#pragma unroll
        for (int mt = 0; mt < MT; ++mt) {
            fa[mt].h[0] = *(const v8bf*)(aP[mt] + k0 + 16 * h);
            fa[mt].h[1] = *(const v8bf*)(aP[mt] + k0 + 16 * h + 8);
        }
#pragma unroll
        for (int nt = 0; nt < NT; ++nt)
            fb[nt] = *(const v16bf*)(wP[nt] + k0);
#pragma unroll
        for (int mt = 0; mt < MT; ++mt)
#pragma unroll
            for (int nt = 0; nt < NT; ++nt)
                acc[mt][nt] = wmma_bf16(fa[mt].v, fb[nt], acc[mt][nt]);
    }

    // D layout: lane (h,q): col = q, rows = 8*h + r (r = 0..7)
    const int col = q;
    const int rbase = h * 8;
#pragma unroll
    for (int mt = 0; mt < MT; ++mt) {
#pragma unroll
        for (int nt = 0; nt < NT; ++nt) {
            const int gcol = nStrip * (16 * NT) + nt * 16 + col;
#pragma unroll
            for (int r = 0; r < 8; ++r) {
                const int grow = mTile * (16 * MT) + mt * 16 + rbase + r;
                const float v = acc[mt][nt][r];
                if (EPI == 0) {
                    outF[(size_t)grow * N + gcol] = v;
                } else if (EPI == 1) {
                    if (gcol < DTRANK)
                        outBF[(size_t)grow * DTRANK + gcol] = (bf16)v;
                    else
                        outF[(size_t)grow * 32 + (gcol - DTRANK)] = v;  // [B(16)|C(16)]
                } else if (EPI == 2) {
                    const float x = v + auxF[gcol];
                    outF[(size_t)grow * N + gcol] = (x > 20.0f) ? x : log1pf(__expf(x));
                } else { // EPI == 3
                    outF[(size_t)grow * N + gcol] = auxF[(size_t)grow * N + gcol] + v;
                }
            }
        }
    }
}

// ---------------- depthwise causal conv1d + SiLU -> bf16 ----------------
// thread = (b, l-chunk of 128, d); x lives in xz[:, 0:DINNER]
__global__ __launch_bounds__(256) void k_conv(const float* __restrict__ xz,
                                              const float* __restrict__ cw,
                                              const float* __restrict__ cb,
                                              bf16* __restrict__ xcb) {
    const int idx = blockIdx.x * 256 + threadIdx.x;   // 65536 threads
    const int d    = idx & (DINNER - 1);
    const int rest = idx >> 11;
    const int lc   = rest & 15;                       // 16 chunks of 128
    const int b    = rest >> 4;
    const int l0   = lc * 128;

    const float* xp = xz + (size_t)b * L_DIM * (2 * DINNER) + d;
    const float w0 = cw[d * 4 + 0], w1 = cw[d * 4 + 1],
                w2 = cw[d * 4 + 2], w3 = cw[d * 4 + 3];
    const float bias = cb[d];

    float xm3 = (l0 >= 3) ? xp[(size_t)(l0 - 3) * (2 * DINNER)] : 0.0f;
    float xm2 = (l0 >= 2) ? xp[(size_t)(l0 - 2) * (2 * DINNER)] : 0.0f;
    float xm1 = (l0 >= 1) ? xp[(size_t)(l0 - 1) * (2 * DINNER)] : 0.0f;

    for (int l = l0; l < l0 + 128; ++l) {
        const float xc = xp[(size_t)l * (2 * DINNER)];
        const float a  = w0 * xm3 + w1 * xm2 + w2 * xm1 + w3 * xc + bias;
        const float s  = a / (1.0f + __expf(-a));     // SiLU
        xcb[((size_t)(b * L_DIM + l)) * DINNER + d] = (bf16)s;
        xm3 = xm2; xm2 = xm1; xm1 = xc;
    }
}

// ---------------- selective scan ----------------
// lane = (channel-pair h, state s): 16 lanes per channel, 2 channels per wave.
__global__ __launch_bounds__(256) void k_scan(const float* __restrict__ dt,
                                              const bf16*  __restrict__ xcb,
                                              const float* __restrict__ bc,
                                              const float* __restrict__ xz,
                                              const float* __restrict__ A_log,
                                              const float* __restrict__ Dv,
                                              bf16* __restrict__ ybf) {
    const int lane = threadIdx.x & 31;
    const int warp = (blockIdx.x * 256 + threadIdx.x) >> 5;  // 2048 warps
    const int ch   = warp * 2 + (lane >> 4);                 // 0..4095
    const int b    = ch >> 11;
    const int d    = ch & (DINNER - 1);
    const int s    = lane & 15;

    const float A_s = -__expf(A_log[d * DSTATE + s]);
    const float D_d = Dv[d];
    float hstate = 0.0f;

    for (int l = 0; l < L_DIM; ++l) {
        const size_t tok = (size_t)b * L_DIM + l;
        const float dtv = dt[tok * DINNER + d];
        const float xv  = (float)xcb[tok * DINNER + d];
        const float Bv  = bc[tok * 32 + s];
        const float Cv  = bc[tok * 32 + 16 + s];
        const float dA  = __expf(dtv * A_s);
        hstate = dA * hstate + (dtv * xv) * Bv;
        float yp = hstate * Cv;
        yp += __shfl_xor(yp, 1, 32);
        yp += __shfl_xor(yp, 2, 32);
        yp += __shfl_xor(yp, 4, 32);
        yp += __shfl_xor(yp, 8, 32);
        if (s == 0) {
            const float zv = xz[tok * (2 * DINNER) + DINNER + d];
            const float g  = zv / (1.0f + __expf(-zv));      // silu(z)
            ybf[tok * DINNER + d] = (bf16)((yp + xv * D_d) * g);
        }
    }
}

// ---------------- host launcher ----------------
extern "C" void kernel_launch(void* const* d_in, const int* in_sizes, int n_in,
                              void* d_out, int out_size, void* d_ws, size_t ws_size,
                              hipStream_t stream) {
    const float* hs        = (const float*)d_in[0];
    const float* norm_w    = (const float*)d_in[1];
    const float* in_proj_w = (const float*)d_in[2];
    const float* conv_w    = (const float*)d_in[3];
    const float* conv_b    = (const float*)d_in[4];
    const float* x_proj_w  = (const float*)d_in[5];
    const float* dt_proj_w = (const float*)d_in[6];
    const float* dt_proj_b = (const float*)d_in[7];
    const float* A_log     = (const float*)d_in[8];
    const float* Dv        = (const float*)d_in[9];
    const float* out_proj_w= (const float*)d_in[10];
    float* out = (float*)d_out;

    char* ws = (char*)d_ws;
    auto alloc = [&](size_t bytes) -> void* {
        void* p = (void*)ws;
        ws += (bytes + 255) & ~(size_t)255;
        return p;
    };
    bf16*  hn_bf   = (bf16*) alloc((size_t)TOKENS * DMODEL * 2);
    bf16*  w_in    = (bf16*) alloc((size_t)2 * DINNER * DMODEL * 2);
    bf16*  w_xp    = (bf16*) alloc((size_t)(DTRANK + 2 * DSTATE) * DINNER * 2);
    bf16*  w_dt    = (bf16*) alloc((size_t)DINNER * DTRANK * 2);
    bf16*  w_out   = (bf16*) alloc((size_t)DMODEL * DINNER * 2);
    float* xz      = (float*)alloc((size_t)TOKENS * 2 * DINNER * 4);
    bf16*  xc_bf   = (bf16*) alloc((size_t)TOKENS * DINNER * 2);
    bf16*  dtraw   = (bf16*) alloc((size_t)TOKENS * DTRANK * 2);
    float* bc      = (float*)alloc((size_t)TOKENS * 32 * 4);
    float* dt      = (float*)alloc((size_t)TOKENS * DINNER * 4);
    bf16*  y_bf    = (bf16*) alloc((size_t)TOKENS * DINNER * 2);
    (void)ws_size; (void)in_sizes; (void)n_in; (void)out_size;

    // 1) weight conversions to bf16
    {
        int n;
        n = 2 * DINNER * DMODEL;          k_cvt_bf16<<<(n + 255) / 256, 256, 0, stream>>>(in_proj_w,  w_in,  n);
        n = (DTRANK + 2*DSTATE) * DINNER; k_cvt_bf16<<<(n + 255) / 256, 256, 0, stream>>>(x_proj_w,   w_xp,  n);
        n = DINNER * DTRANK;              k_cvt_bf16<<<(n + 255) / 256, 256, 0, stream>>>(dt_proj_w,  w_dt,  n);
        n = DMODEL * DINNER;              k_cvt_bf16<<<(n + 255) / 256, 256, 0, stream>>>(out_proj_w, w_out, n);
    }

    // 2) RMSNorm -> bf16
    k_rmsnorm<<<TOKENS, 256, 0, stream>>>(hs, norm_w, hn_bf);

    // 3) in_proj GEMM: xz[4096,4096] = hn * W_in^T   (fp32 out), 32x64 wave tile
    {
        int waves = (TOKENS / 32) * ((2 * DINNER) / 64);
        k_gemm<0, 2, 4><<<waves / 8, 256, 0, stream>>>(hn_bf, w_in, TOKENS, 2 * DINNER, DMODEL,
                                                       xz, nullptr, nullptr);
    }

    // 4) depthwise conv + SiLU -> xc_bf
    k_conv<<<(B_DIM * 16 * DINNER) / 256, 256, 0, stream>>>(xz, conv_w, conv_b, xc_bf);

    // 5) x_proj GEMM: [4096,96] -> dt_raw(bf16) + B/C(fp32), 32x32 wave tile
    {
        int waves = (TOKENS / 32) * ((DTRANK + 2 * DSTATE) / 32);
        k_gemm<1, 2, 2><<<waves / 8, 256, 0, stream>>>(xc_bf, w_xp, TOKENS, DTRANK + 2 * DSTATE, DINNER,
                                                       bc, dtraw, nullptr);
    }

    // 6) dt GEMM: dt[4096,2048] = softplus(dt_raw * W_dt^T + b), 32x64 wave tile
    {
        int waves = (TOKENS / 32) * (DINNER / 64);
        k_gemm<2, 2, 4><<<waves / 8, 256, 0, stream>>>(dtraw, w_dt, TOKENS, DINNER, DTRANK,
                                                       dt, nullptr, dt_proj_b);
    }

    // 7) selective scan -> y(bf16), fused +x*D and *silu(z)
    k_scan<<<(B_DIM * DINNER * DSTATE) / 256, 256, 0, stream>>>(dt, xc_bf, bc, xz, A_log, Dv, y_bf);

    // 8) out_proj GEMM + residual -> d_out (fp32), 32x64 wave tile
    {
        int waves = (TOKENS / 32) * (DMODEL / 64);
        k_gemm<3, 2, 4><<<waves / 8, 256, 0, stream>>>(y_bf, w_out, TOKENS, DMODEL, DINNER,
                                                       out, nullptr, hs);
    }
}
